// EnhancedIASNHFramework_75814762709679
// MI455X (gfx1250) — matrix operand order repaired and checked by reading, not compiled, not run on hardware
//
#include <hip/hip_runtime.h>

#define B_N 32768
#define BN_INV_C 0.99999500003749969f
#define INFF __builtin_inff()

typedef unsigned short u16;
typedef __attribute__((ext_vector_type(16))) __bf16 v16bf;
typedef __attribute__((ext_vector_type(8)))  float  v8f;
typedef __attribute__((ext_vector_type(4)))  unsigned int u32x4;

union ABfrag { u32x4 q[2]; v16bf v; };

__device__ __forceinline__ u16 f2bf(float f){
  union { __bf16 h; u16 u; } c;
  c.h = (__bf16)f;           // native v_cvt f32->bf16 (RNE)
  return c.u;
}

__device__ __forceinline__ float wsum(float v){
#pragma unroll
  for (int m = 16; m >= 1; m >>= 1) v += __shfl_xor(v, m, 32);
  return v;
}
__device__ __forceinline__ float wmin(float v){
#pragma unroll
  for (int m = 16; m >= 1; m >>= 1) v = fminf(v, __shfl_xor(v, m, 32));
  return v;
}
__device__ __forceinline__ float wmaxr(float v){
#pragma unroll
  for (int m = 16; m >= 1; m >>= 1) v = fmaxf(v, __shfl_xor(v, m, 32));
  return v;
}

// ---------------------------------------------------------------------------
// Generic WMMA GEMM: D[M,Npad] = act(A[M,K] * Wt[K,Npad] + bias) (*scale+shift)
// A, Wt bf16 (u16).  Each wave computes a 16 x (16*NT) output tile: one shared
// A fragment feeds NT independent accumulators per K-step, so the bf16
// WMMA->WMMA RAW hazard is hidden by unrelated WMMAs instead of v_nops.
// ---------------------------------------------------------------------------
template<int K, int ACT, bool OUTF32, int NT>
__global__ __launch_bounds__(256) void gemm_bf16(
    const u16* __restrict__ A, const u16* __restrict__ Wt,
    const float* __restrict__ bias, const float* __restrict__ scale,
    const float* __restrict__ shift, void* __restrict__ Dv,
    int Npad, int ldd, int dcol0, int tiles)
{
  int wid = blockIdx.x * (blockDim.x >> 5) + (threadIdx.x >> 5);
  if (wid >= tiles) return;
  int tiles_n = Npad / (16 * NT);
  int mtile = wid / tiles_n;
  int ngrp  = wid - mtile * tiles_n;
  int lane = threadIdx.x & 31;
  int hlf  = lane >> 4;
  int mrow = mtile * 16 + (lane & 15);
  const u16* arow = A + (size_t)mrow * K;
  const u16* bbase = Wt + (size_t)(lane & 15) + ngrp * (16 * NT);

  v8f acc[NT];
#pragma unroll
  for (int j = 0; j < NT; ++j) acc[j] = (v8f){0.f,0.f,0.f,0.f,0.f,0.f,0.f,0.f};

#pragma unroll
  for (int k0 = 0; k0 < K; k0 += 32){
    ABfrag a;
    // A 16x32 bf16 layout: lanes 0-15 rows; elems 0..7 -> K=k0+8*hlf+0..7,
    // elems 8..15 -> K=k0+16+8*hlf+0..7  (two contiguous 16B loads)
    a.q[0] = *(const u32x4*)(arow + k0 + 8 * hlf);
    a.q[1] = *(const u32x4*)(arow + k0 + 16 + 8 * hlf);
    // B 32x16 bf16 layout: lanes 0-15 cols N; same K order as A
    const u16* brow = bbase + (size_t)k0 * Npad;
#pragma unroll
    for (int j = 0; j < NT; ++j){
      ABfrag b;
      u16 bt[16];
#pragma unroll
      for (int s = 0; s < 16; ++s){
        int kk = (s < 8) ? (8 * hlf + s) : (16 + 8 * hlf + (s - 8));
        bt[s] = brow[(size_t)kk * Npad + 16 * j];
      }
      __builtin_memcpy(&b, bt, 32);
      acc[j] = __builtin_amdgcn_wmma_f32_16x16x32_bf16(
          false, a.v, false, b.v, (short)0, acc[j], false, false);
    }
  }

  int nb = ngrp * (16 * NT) + (lane & 15);
#pragma unroll
  for (int j = 0; j < NT; ++j){
    int n = nb + 16 * j;
    float bv = bias[n];
    float sc = (ACT == 2) ? scale[n] : 0.f;
    float sh = (ACT == 2) ? shift[n] : 0.f;
#pragma unroll
    for (int r = 0; r < 8; ++r){
      int m = mtile * 16 + r + (hlf << 3);  // C/D layout: lanes16-31 -> M+8
      float v = acc[j][r] + bv;
      if (ACT >= 1) v = fmaxf(v, 0.f);
      if (ACT == 2) v = v * sc + sh;
      size_t off = (size_t)m * ldd + dcol0 + n;
      if (OUTF32) ((float*)Dv)[off] = v;
      else        ((u16*)Dv)[off]   = f2bf(v);
    }
  }
}

// ---------------------------------------------------------------------------
// Weight prep: W (N x K, f32) -> Wt (K x Npad, bf16, zero padded), bias/bn pad
// ---------------------------------------------------------------------------
__global__ void prep_layer(const float* __restrict__ W, const float* __restrict__ b,
                           const float* __restrict__ g, const float* __restrict__ be,
                           int N, int K, int Npad,
                           u16* __restrict__ Wt, float* __restrict__ biasP,
                           float* __restrict__ scP, float* __restrict__ shP)
{
  int i = blockIdx.x * blockDim.x + threadIdx.x;
  int total = K * Npad;
  if (i < total){
    int k = i / Npad, n = i - k * Npad;
    float v = (n < N) ? W[(size_t)n * K + k] : 0.f;
    Wt[i] = f2bf(v);
  }
  if (i < Npad){
    biasP[i] = (i < N) ? b[i] : 0.f;
    if (scP){
      scP[i] = (i < N) ? g[i] * BN_INV_C : 0.f;
      shP[i] = (i < N) ? be[i] : 0.f;
    }
  }
}

// ---------------------------------------------------------------------------
// Fused conv stack: (B,1,96) -> conv7 relu bn pool -> conv5 relu bn pool ->
// conv3 relu bn mean -> h (B,128) bf16.  One 128-thread block per sample.
// ---------------------------------------------------------------------------
__global__ __launch_bounds__(128) void conv_stack_kernel(
    const float* __restrict__ x,
    const float* __restrict__ cw1, const float* __restrict__ cb1,
    const float* __restrict__ g1,  const float* __restrict__ be1,
    const float* __restrict__ cw2, const float* __restrict__ cb2,
    const float* __restrict__ g2,  const float* __restrict__ be2,
    const float* __restrict__ cw3, const float* __restrict__ cb3,
    const float* __restrict__ g3,  const float* __restrict__ be3,
    u16* __restrict__ hbuf)
{
  __shared__ float sx[102];              // x padded by 3 each side
  __shared__ float b1s[32][48];          // pooled conv1 output
  __shared__ float b2s[64][24];          // pooled conv2 output
  __shared__ float w2s[64 * 32 * 5];     // conv2 weights (40KB)
  int tid = threadIdx.x;
  size_t b = blockIdx.x;

  for (int i = tid; i < 102; i += 128)
    sx[i] = (i >= 3 && i < 99) ? x[b * 96 + (i - 3)] : 0.f;
  for (int i = tid; i < 10240; i += 128) w2s[i] = cw2[i];
  __syncthreads();

  { // conv1: k=7 SAME, relu, bn, pool2 -> (32,48)
    int co = tid & 31, grp = tid >> 5;
    float wv[7];
#pragma unroll
    for (int k = 0; k < 7; k++) wv[k] = cw1[co * 7 + k];
    float bb = cb1[co], sc = g1[co] * BN_INV_C, sh = be1[co];
    for (int tp = grp * 12; tp < grp * 12 + 12; ++tp){
      float a0 = bb, a1 = bb;
      int t0 = 2 * tp;
#pragma unroll
      for (int k = 0; k < 7; k++){
        a0 += sx[t0 + k] * wv[k];
        a1 += sx[t0 + 1 + k] * wv[k];
      }
      float v0 = fmaxf(a0, 0.f) * sc + sh;
      float v1 = fmaxf(a1, 0.f) * sc + sh;
      b1s[co][tp] = fmaxf(v0, v1);
    }
  }
  __syncthreads();

  { // conv2: k=5 SAME over len48, relu, bn, pool2 -> (64,24)
    int co = tid & 63, grp = tid >> 6;
    float bb = cb2[co], sc = g2[co] * BN_INV_C, sh = be2[co];
    for (int tp = grp * 12; tp < grp * 12 + 12; ++tp){
      float a0 = bb, a1 = bb;
      int t0 = 2 * tp;
      for (int ci = 0; ci < 32; ++ci){
        const float* wr = &w2s[(co * 32 + ci) * 5];
        const float* row = b1s[ci];
#pragma unroll
        for (int k = 0; k < 5; k++){
          int i0 = t0 - 2 + k;
          float w = wr[k];
          if (i0 >= 0 && i0 < 48) a0 += row[i0] * w;
          int i1 = i0 + 1;
          if (i1 >= 0 && i1 < 48) a1 += row[i1] * w;
        }
      }
      float v0 = fmaxf(a0, 0.f) * sc + sh;
      float v1 = fmaxf(a1, 0.f) * sc + sh;
      b2s[co][tp] = fmaxf(v0, v1);
    }
  }
  __syncthreads();

  { // conv3: k=3 SAME over len24, relu, bn, mean -> h[128]
    int co = tid;
    float acc[24];
    float bb = cb3[co];
#pragma unroll
    for (int t = 0; t < 24; t++) acc[t] = bb;
    for (int ci = 0; ci < 64; ++ci){
      const float* wr = cw3 + (size_t)(co * 64 + ci) * 3;
      float w0 = wr[0], w1 = wr[1], w2 = wr[2];
      const float* row = b2s[ci];
#pragma unroll
      for (int t = 0; t < 24; t++){
        float s = row[t] * w1;
        if (t > 0)  s += row[t - 1] * w0;
        if (t < 23) s += row[t + 1] * w2;
        acc[t] += s;
      }
    }
    float sc = g3[co] * BN_INV_C, sh = be3[co];
    float m = 0.f;
#pragma unroll
    for (int t = 0; t < 24; t++) m += fmaxf(acc[t], 0.f) * sc + sh;
    hbuf[b * 128 + co] = f2bf(m * (1.f / 24.f));
  }
}

// ---------------------------------------------------------------------------
// Statistics features: one wave32 per sample, 4 waves per block.
// ---------------------------------------------------------------------------
__device__ void bitonic_sort(float* S, int n, int lane){
  for (int ksz = 2; ksz <= n; ksz <<= 1)
    for (int j = ksz >> 1; j > 0; j >>= 1){
      __syncthreads();
      for (int i = lane; i < n; i += 32){
        int ixj = i ^ j;
        if (ixj > i){
          float a = S[i], c = S[ixj];
          bool up = ((i & ksz) == 0);
          if (up ? (a > c) : (a < c)){ S[i] = c; S[ixj] = a; }
        }
      }
    }
  __syncthreads();
}

__global__ __launch_bounds__(128) void stats_kernel(
    const float* __restrict__ x, u16* __restrict__ statb)
{
  __shared__ float X[4][96];
  __shared__ float S[4][128];
  __shared__ float FM[4][64];
  __shared__ int   H[4][10];
  __shared__ float F[4][32];
  int tid = threadIdx.x;
  int w = tid >> 5, lane = tid & 31;
  size_t b = (size_t)blockIdx.x * 4 + w;
  const float* xr = x + b * 96;
  for (int i = lane; i < 96; i += 32) X[w][i] = xr[i];
  if (lane < 10) H[w][lane] = 0;
  __syncthreads();
  const float* Xw = X[w];

  // mean / min / max
  float s = 0, mn = INFF, mx = -INFF;
  for (int i = lane; i < 96; i += 32){ float v = Xw[i]; s += v; mn = fminf(mn, v); mx = fmaxf(mx, v); }
  s = wsum(s); mn = wmin(mn); mx = wmaxr(mx);
  float mean = s * (1.f / 96.f);

  // central moments + slope numerator
  float m2 = 0, m3 = 0, m4 = 0, st = 0;
  for (int i = lane; i < 96; i += 32){
    float d = Xw[i] - mean, dd = d * d;
    m2 += dd; m3 += dd * d; m4 += dd * dd;
    st += Xw[i] * ((float)i - 47.5f);
  }
  m2 = wsum(m2) * (1.f / 96.f);
  m3 = wsum(m3) * (1.f / 96.f);
  m4 = wsum(m4) * (1.f / 96.f);
  st = wsum(st);
  float var  = m2;
  float stdv = sqrtf(fmaxf(m2, 0.f));
  float skew = m3 / (m2 * sqrtf(m2));
  float kurt = m4 / (m2 * m2) - 3.f;
  float slope = st / 73720.f;                 // sum((i-47.5)^2) = (96^3-96)/12

  // first / second differences
  float Sd = 0, Sdd = 0, Sad = 0;
  for (int i = lane; i < 95; i += 32){ float d = Xw[i+1] - Xw[i]; Sd += d; Sdd += d * d; Sad += fabsf(d); }
  Sd = wsum(Sd); Sdd = wsum(Sdd); Sad = wsum(Sad);
  float md = Sd / 95.f;
  float std_d = sqrtf(fmaxf(Sdd / 95.f - md * md, 0.f));
  float sad = Sad, madf = Sad / 95.f;
  float S2 = 0, S22 = 0;
  for (int i = lane; i < 94; i += 32){ float dd = Xw[i+2] - 2.f * Xw[i+1] + Xw[i]; S2 += dd; S22 += dd * dd; }
  S2 = wsum(S2); S22 = wsum(S22);
  float md2 = S2 / 94.f;
  float std_d2 = sqrtf(fmaxf(S22 / 94.f - md2 * md2, 0.f));
  float nonlin = std_d2 / (std_d + 1e-8f);
  float trend  = std_d / (stdv + 1e-8f);
  float vol    = std_d / (fabsf(mean) + 1e-8f);
  float cv     = stdv  / (fabsf(mean) + 1e-8f);

  // autocorrelations
  auto ac = [&](int k) -> float {
    float Sa = 0, Sc = 0, Saa = 0, Scc = 0, Sac = 0;
    int n2 = 96 - k;
    for (int i = lane; i < n2; i += 32){
      float a = Xw[i], c = Xw[i + k];
      Sa += a; Sc += c; Saa += a * a; Scc += c * c; Sac += a * c;
    }
    Sa = wsum(Sa); Sc = wsum(Sc); Saa = wsum(Saa); Scc = wsum(Scc); Sac = wsum(Sac);
    float nf2 = (float)n2;
    float num = Sac - Sa * Sc / nf2;
    float da = Saa - Sa * Sa / nf2, dc = Scc - Sc * Sc / nf2;
    return num / sqrtf(da * dc);
  };
  float ac1 = ac(1), ac7 = ac(7), ac12 = ac(12), ac24 = ac(24), ac48 = ac(48);

  // subsampled variances (redundant per-lane, tiny)
  float m24 = 0; for (int i = 0; i < 4; i++) m24 += Xw[i * 24]; m24 *= 0.25f;
  float v24 = 0; for (int i = 0; i < 4; i++){ float d = Xw[i * 24] - m24; v24 += d * d; } v24 *= 0.25f;
  float s24 = v24 / (var + 1e-8f);
  float m12 = 0; for (int i = 0; i < 8; i++) m12 += Xw[i * 12]; m12 *= 0.125f;
  float v12 = 0; for (int i = 0; i < 8; i++){ float d = Xw[i * 12] - m12; v12 += d * d; } v12 *= 0.125f;
  float s12 = v12 / (var + 1e-8f);
  float season = fmaxf(s24, s12);

  // stationarity
  float sA = 0, sAA = 0, sB = 0, sBB = 0;
  for (int i = lane; i < 48; i += 32){
    float v = Xw[i];      sA += v; sAA += v * v;
    float u = Xw[i + 48]; sB += u; sBB += u * u;
  }
  sA = wsum(sA); sAA = wsum(sAA); sB = wsum(sB); sBB = wsum(sBB);
  float mA = sA / 48.f, mB = sB / 48.f;
  float st1 = sqrtf(fmaxf(sAA / 48.f - mA * mA, 0.f));
  float st2 = sqrtf(fmaxf(sBB / 48.f - mB * mB, 0.f));
  float stat_ = fabsf(st1 - st2) / (stdv + 1e-8f);

  // histogram entropy
  float width = mx - mn;
  float swd = (width > 0.f) ? width : 1.f;
  for (int i = lane; i < 96; i += 32){
    int idx = (int)floorf((Xw[i] - mn) / swd * 10.f);
    idx = min(max(idx, 0), 9);
    atomicAdd(&H[w][idx], 1);
  }
  __syncthreads();
  float ent = 0.f;
  for (int i = 0; i < 10; i++){
    float p = (float)H[w][i] * (1.f / 96.f);
    ent -= p * logf(p + 1e-8f);
  }

  // DFT magnitudes (48 bins), spectral centroid + rolloff percentile
  const float W0 = -6.28318530717958647692f / 96.f;
  for (int f = lane; f < 48; f += 32){
    float re = 0, im = 0, wf = W0 * (float)f;
    for (int t = 0; t < 96; t++){
      float ang = wf * (float)t;
      re += Xw[t] * cosf(ang);
      im += Xw[t] * sinf(ang);
    }
    FM[w][f] = sqrtf(re * re + im * im);
  }
  for (int i = 48 + lane; i < 64; i += 32) FM[w][i] = INFF;
  __syncthreads();
  float Sf = 0, Sff = 0;
  for (int f = lane; f < 48; f += 32){ float v = FM[w][f]; Sf += v; Sff += v * (float)f; }
  Sf = wsum(Sf); Sff = wsum(Sff);
  float sc_ = Sff / (Sf + 1e-8f);
  bitonic_sort(FM[w], 64, lane);
  float sr = FM[w][39] * 0.05f + FM[w][40] * 0.95f;   // 85th pct of 48

  // order statistics of x
  for (int i = lane; i < 128; i += 32) S[w][i] = (i < 96) ? Xw[i] : INFF;
  bitonic_sort(S[w], 128, lane);
  const float* Sw = S[w];
  float med = (Sw[47] + Sw[48]) * 0.5f;
  float q25 = Sw[23] * 0.25f + Sw[24] * 0.75f;
  float q75 = Sw[71] * 0.75f + Sw[72] * 0.25f;
  float q10 = (Sw[9]  + Sw[10]) * 0.5f;
  float q90 = (Sw[85] + Sw[86]) * 0.5f;
  float iqr = q75 - q25;
  float lo = q25 - 1.5f * iqr, hi = q75 + 1.5f * iqr;
  float oc = 0;
  for (int i = lane; i < 96; i += 32){ float v = Xw[i]; if (v < lo || v > hi) oc += 1.f; }
  oc = wsum(oc) * (1.f / 96.f);

  if (lane == 0){
    float* Fp = F[w];
    Fp[0]=mean; Fp[1]=stdv; Fp[2]=mn; Fp[3]=mx; Fp[4]=med; Fp[5]=trend; Fp[6]=slope;
    Fp[7]=ac1; Fp[8]=ac7; Fp[9]=ac12; Fp[10]=ac24; Fp[11]=ac48; Fp[12]=cv; Fp[13]=iqr;
    Fp[14]=s24; Fp[15]=s12; Fp[16]=skew; Fp[17]=kurt; Fp[18]=ent; Fp[19]=stat_; Fp[20]=oc;
    Fp[21]=nonlin; Fp[22]=sc_; Fp[23]=sr; Fp[24]=vol; Fp[25]=ac1; Fp[26]=season; Fp[27]=96.f;
    Fp[28]=sad; Fp[29]=madf; Fp[30]=q10; Fp[31]=q90;
  }
  __syncthreads();
  float v = F[w][lane];
  v = (v - v == 0.f) ? v : 0.f;   // isfinite
  statb[b * 32 + lane] = f2bf(v);
}

// ---------------------------------------------------------------------------
// Heads: softmax(logits/clip(temp)) (12-wide) + sigmoid(conf)
// ---------------------------------------------------------------------------
__global__ __launch_bounds__(256) void head_kernel(
    const float* __restrict__ logits, const float* __restrict__ confb,
    const float* __restrict__ temp_p, float* __restrict__ out)
{
  int b = blockIdx.x * 256 + threadIdx.x;
  if (b >= B_N) return;
  float t = temp_p[0];
  t = fminf(fmaxf(t, 0.1f), 10.f);
  float l[12], mx = -INFF;
#pragma unroll
  for (int j = 0; j < 12; j++){ l[j] = logits[(size_t)b * 16 + j] / t; mx = fmaxf(mx, l[j]); }
  float s = 0;
#pragma unroll
  for (int j = 0; j < 12; j++){ l[j] = expf(l[j] - mx); s += l[j]; }
  float inv = 1.f / s;
#pragma unroll
  for (int j = 0; j < 12; j++) out[(size_t)b * 12 + j] = l[j] * inv;
  float c = confb[(size_t)b * 16];
  out[(size_t)B_N * 12 + b] = 1.f / (1.f + expf(-c));
}

// ---------------------------------------------------------------------------
// Host orchestration
// ---------------------------------------------------------------------------
struct LayerDef { int Widx, bidx, gidx, beidx; size_t woff, boff; int N, K, Npad; };
static const LayerDef LD[16] = {
  {13,14,-1,-1, 0,0, 128,128,128},   // 0 fcw        act0
  {15,16,17,18, 0,0, 128, 32,128},   // 1 sw1        relu+bn
  {19,20,-1,-1, 0,0,  64,128, 64},   // 2 sw2        relu
  {21,22,-1,-1, 0,0, 128, 64,128},   // 3 sw3        act0
  {23,24,25,26, 0,0, 256,256,256},   // 4 fw1        relu+bn
  {27,28,-1,-1, 0,0, 128,256,128},   // 5 fw2        relu
  {29,30,-1,-1, 0,0, 128,128,128},   // 6 fw3        act0
  {31,32,-1,-1, (size_t)256*128,256, 128,128,128},  // 7 aw_in[2FD:]  act0
  {33,34,-1,-1, 0,0, 128,128,128},   // 8 aw_out     act0
  {35,36,37,38, 0,0, 512,128,512},   // 9 mw1        relu+bn
  {39,40,41,42, 0,0, 256,512,256},   // 10 mw2       relu+bn
  {43,44,-1,-1, 0,0, 128,256,128},   // 11 mw3       relu
  {45,46,-1,-1, 0,0,  12,128, 16},   // 12 mw4       act0, f32 out
  {47,48,49,50, 0,0, 256,128,256},   // 13 cfw1      relu+bn
  {51,52,-1,-1, 0,0,  64,256, 64},   // 14 cfw2      relu
  {53,54,-1,-1, 0,0,   1, 64, 16},   // 15 cfw3      act0, f32 out
};

extern "C" void kernel_launch(void* const* d_in, const int* in_sizes, int n_in,
                              void* d_out, int out_size, void* d_ws, size_t ws_size,
                              hipStream_t stream) {
  (void)in_sizes; (void)n_in; (void)out_size; (void)ws_size;
  const int Bn = B_N;
  const float* X = (const float*)d_in[0];

  char* wsp = (char*)d_ws;
  size_t off = 0;
  auto alloc = [&](size_t bytes) -> char* {
    char* p = wsp + off;
    off = (off + bytes + 255) & ~(size_t)255;
    return p;
  };

  // activation buffers (bf16 unless noted)
  u16* hbuf    = (u16*)alloc((size_t)Bn * 128 * 2);
  u16* statbuf = (u16*)alloc((size_t)Bn * 32 * 2);
  u16* fbuf    = (u16*)alloc((size_t)Bn * 256 * 2);   // concat [conv_feat | stat_feat]
  u16* sbuf1   = (u16*)alloc((size_t)Bn * 128 * 2);
  u16* sbuf2   = (u16*)alloc((size_t)Bn * 64 * 2);
  u16* fbuf2   = (u16*)alloc((size_t)Bn * 256 * 2);
  u16* fbuf3   = (u16*)alloc((size_t)Bn * 128 * 2);
  u16* fusedb  = (u16*)alloc((size_t)Bn * 128 * 2);
  u16* vbuf    = (u16*)alloc((size_t)Bn * 128 * 2);
  u16* attbuf  = (u16*)alloc((size_t)Bn * 128 * 2);
  u16* mbuf1   = (u16*)alloc((size_t)Bn * 512 * 2);
  u16* mbuf2   = (u16*)alloc((size_t)Bn * 256 * 2);
  u16* mbuf3   = (u16*)alloc((size_t)Bn * 128 * 2);
  float* logitb= (float*)alloc((size_t)Bn * 16 * 4);
  u16* cbuf1   = (u16*)alloc((size_t)Bn * 256 * 2);
  u16* cbuf2   = (u16*)alloc((size_t)Bn * 64 * 2);
  float* confb = (float*)alloc((size_t)Bn * 16 * 4);

  // weight prep
  u16* Wt[16]; float* bp[16]; float* scp[16]; float* shp[16];
  for (int l = 0; l < 16; l++){
    const LayerDef& d = LD[l];
    Wt[l] = (u16*)alloc((size_t)d.K * d.Npad * 2);
    bp[l] = (float*)alloc((size_t)d.Npad * 4);
    bool bn = d.gidx >= 0;
    scp[l] = bn ? (float*)alloc((size_t)d.Npad * 4) : nullptr;
    shp[l] = bn ? (float*)alloc((size_t)d.Npad * 4) : nullptr;
    const float* Wp = (const float*)d_in[d.Widx] + d.woff;
    const float* Bp = (const float*)d_in[d.bidx] + d.boff;
    int total = d.K * d.Npad;
    int work = total > d.Npad ? total : d.Npad;
    int blocks = (work + 255) / 256;
    prep_layer<<<blocks, 256, 0, stream>>>(
        Wp, Bp,
        bn ? (const float*)d_in[d.gidx]  : nullptr,
        bn ? (const float*)d_in[d.beidx] : nullptr,
        d.N, d.K, d.Npad, Wt[l], bp[l], scp[l], shp[l]);
  }

  // conv stack + stat features
  conv_stack_kernel<<<Bn, 128, 0, stream>>>(
      X,
      (const float*)d_in[1], (const float*)d_in[2], (const float*)d_in[3], (const float*)d_in[4],
      (const float*)d_in[5], (const float*)d_in[6], (const float*)d_in[7], (const float*)d_in[8],
      (const float*)d_in[9], (const float*)d_in[10],(const float*)d_in[11],(const float*)d_in[12],
      hbuf);
  stats_kernel<<<Bn / 4, 128, 0, stream>>>(X, statbuf);

#define GEMM(Kv, ACTv, OUTF, NTv, Ap, l, Dp, lddv, dcolv) do {                     \
    int tiles_ = (Bn / 16) * (LD[l].Npad / (16 * NTv));                            \
    int blocks_ = (tiles_ + 7) / 8;                                                \
    gemm_bf16<Kv, ACTv, OUTF, NTv><<<blocks_, 256, 0, stream>>>(                   \
        (const u16*)(Ap), Wt[l], bp[l], scp[l], shp[l], (void*)(Dp),               \
        LD[l].Npad, lddv, dcolv, tiles_);                                          \
  } while (0)

  GEMM(128, 0, false, 4, hbuf,    0,  fbuf,   256, 0);    // conv_feat -> fbuf[:, :128]
  GEMM(32,  2, false, 4, statbuf, 1,  sbuf1,  128, 0);    // relu+bn
  GEMM(128, 1, false, 4, sbuf1,   2,  sbuf2,   64, 0);    // relu
  GEMM(64,  0, false, 4, sbuf2,   3,  fbuf,   256, 128);  // stat_feat -> fbuf[:, 128:]
  GEMM(256, 2, false, 4, fbuf,    4,  fbuf2,  256, 0);    // relu+bn
  GEMM(256, 1, false, 4, fbuf2,   5,  fbuf3,  128, 0);    // relu
  GEMM(128, 0, false, 4, fbuf3,   6,  fusedb, 128, 0);    // fused
  GEMM(128, 0, false, 4, fusedb,  7,  vbuf,   128, 0);    // v
  GEMM(128, 0, false, 4, vbuf,    8,  attbuf, 128, 0);    // att
  GEMM(128, 2, false, 4, attbuf,  9,  mbuf1,  512, 0);    // relu+bn
  GEMM(512, 2, false, 4, mbuf1,   10, mbuf2,  256, 0);    // relu+bn
  GEMM(256, 1, false, 4, mbuf2,   11, mbuf3,  128, 0);    // relu
  GEMM(128, 0, true,  1, mbuf3,   12, logitb,  16, 0);    // logits (f32)
  GEMM(128, 2, false, 4, attbuf,  13, cbuf1,  256, 0);    // relu+bn
  GEMM(256, 1, false, 4, cbuf1,   14, cbuf2,   64, 0);    // relu
  GEMM(64,  0, true,  1, cbuf2,   15, confb,   16, 0);    // conf pre-sigmoid (f32)
#undef GEMM

  head_kernel<<<Bn / 256, 256, 0, stream>>>(logitb, confb, (const float*)d_in[55],
                                            (float*)d_out);
}
